// LSTM_21157008900096
// MI455X (gfx1250) — matrix operand order repaired and checked by reading, not compiled, not run on hardware
//
#include <hip/hip_runtime.h>
#include <hip/hip_bf16.h>

// ---------------------------------------------------------------------------
// LSTM (packed-sequence, eval-mode recurrent dropout = scalar 0.7 scale)
// T=2048, B=64, D=256, H=256, gates 4H=1024.
//
// Plan (MI455X / gfx1250, wave32, WMMA):
//  prep   : bf16-ify x*p, W, U                           (bandwidth bound)
//  proj   : G[T*B,1024] = (x*p)@W^T + (b_ih+b_hh)        (WMMA bf16, parallel,
//           4 independent accumulator chains per wave to hide the
//           WMMA->WMMA RAW hazard window)
//  recur  : 2048 sequential steps on ONE WGP. h,c live in LDS (160KB of the
//           320KB CDNA5 LDS), G streamed from HBM into WMMA accumulators
//           (global_prefetch_b8 one step ahead), U hot in L2.
//           32 waves x (2 M-tiles x 4 gate tiles x 8 K-chunks)
//           = 64 v_wmma_f32_16x16x32_bf16 per wave per step, issued as
//           8 independent accumulator chains (no hazard stalls).
// ---------------------------------------------------------------------------

#define TT 2048
#define BB 64
#define DD 256
#define HH 256
#define NG 1024            // 4*H
#define PKEEP 0.7f         // 1 - dropout

typedef __bf16 bf16_t;
typedef __attribute__((ext_vector_type(16))) __bf16 v16bf;
typedef __attribute__((ext_vector_type(8)))  __bf16 v8bf;
typedef __attribute__((ext_vector_type(8)))  float  v8f;

static __device__ __forceinline__ float sigf(float x) {
    return 1.0f / (1.0f + __expf(-x));
}

// ---------------------------------------------------------------------------
// Kernel 0: convert inputs to bf16 (x is pre-scaled by PKEEP; h scaled later).
// ---------------------------------------------------------------------------
extern "C" __global__ __launch_bounds__(256)
void lstm_prep(const float* __restrict__ x, const float* __restrict__ W,
               const float* __restrict__ U,
               bf16_t* __restrict__ Xbf, bf16_t* __restrict__ Wbf,
               bf16_t* __restrict__ Ubf) {
    const size_t stride = (size_t)gridDim.x * blockDim.x;
    const size_t base   = (size_t)blockIdx.x * blockDim.x + threadIdx.x;
    const size_t nx = (size_t)TT * BB * DD;     // 33.5M
    const size_t nw = (size_t)NG * DD;          // 262K
    for (size_t i = base; i < nx; i += stride)
        Xbf[i] = (bf16_t)(x[i] * PKEEP);
    for (size_t i = base; i < nw; i += stride) {
        Wbf[i] = (bf16_t)W[i];
        Ubf[i] = (bf16_t)U[i];
    }
}

// ---------------------------------------------------------------------------
// Kernel 1: G = Xbf @ W^T + (b_ih + b_hh).  One 16-row M-tile per block,
// 8 waves/block, each wave owns 8 N-tiles processed as 2 groups of 4
// concurrent accumulator chains (hides the bf16 WMMA RAW hazard: distance-4
// reuse of each accumulator).  A (x rows) hoisted once across all N-tiles.
//
// WMMA bf16 operand layouts used here (contiguous K, not the ISA table order):
//   A 16x32: lane<16 -> M=lane; 16 contiguous K at b1,
//            b1 = kc*32 + (lane>>4)*16
//   B 32x16: lane col N = lane&15, 16 contiguous K at kc*32 + (lane>>4)*16
//   C/D 16x16 f32: row = r + 8*(lane>=16), col = lane&15
// ---------------------------------------------------------------------------
extern "C" __global__ __launch_bounds__(256)
void lstm_input_proj(const bf16_t* __restrict__ Xbf,   // [T*B, D]
                     const bf16_t* __restrict__ Wbf,   // [NG, D]
                     const float*  __restrict__ b_ih,  // [NG]
                     const float*  __restrict__ b_hh,  // [NG]
                     float* __restrict__ G) {          // [T*B, NG]
    const int tid   = threadIdx.x;
    const int lane  = tid & 31;
    const int wave  = tid >> 5;        // 0..7
    const int l15   = lane & 15;
    const int lhalf = lane >> 4;       // 0/1
    const int mt    = blockIdx.x;      // 0 .. T*B/16 - 1

    // Hoist all 8 K-chunks of the A tile (this wave's 16 x-rows) once.
    v16bf A[8];
    const bf16_t* arow = Xbf + (size_t)(mt * 16 + l15) * DD;
#pragma unroll
    for (int kc = 0; kc < 8; ++kc) {
        const v8bf x1 = *(const v8bf*)(arow + kc * 32 + lhalf * 16);
        const v8bf x2 = *(const v8bf*)(arow + kc * 32 + lhalf * 16 + 8);
#pragma unroll
        for (int i = 0; i < 8; ++i) { A[kc][i] = x1[i]; A[kc][i + 8] = x2[i]; }
    }

    // 8 N-tiles per wave, 4 concurrent accumulator chains at a time.
#pragma unroll
    for (int grp = 0; grp < 2; ++grp) {
        const int nt0 = wave * 8 + grp * 4;
        v8f acc[4];
        const bf16_t* bcol[4];
#pragma unroll
        for (int q = 0; q < 4; ++q) {
            const int col = (nt0 + q) * 16 + l15;
            const float bias = b_ih[col] + b_hh[col];
#pragma unroll
            for (int r = 0; r < 8; ++r) acc[q][r] = bias;
            bcol[q] = Wbf + (size_t)col * DD + lhalf * 16;
        }
#pragma unroll
        for (int kc = 0; kc < 8; ++kc) {
#pragma unroll
            for (int q = 0; q < 4; ++q) {   // 4 independent WMMA chains
                const v16bf bm = *(const v16bf*)(bcol[q] + kc * 32);
                acc[q] = __builtin_amdgcn_wmma_f32_16x16x32_bf16(
                    false, A[kc], false, bm, (short)0, acc[q], false, false);
            }
        }
#pragma unroll
        for (int q = 0; q < 4; ++q) {
            const int col = (nt0 + q) * 16 + l15;
            float* gp = G + (size_t)(mt * 16 + lhalf * 8) * NG + col;
#pragma unroll
            for (int r = 0; r < 8; ++r) gp[(size_t)r * NG] = acc[q][r];
        }
    }
}

// ---------------------------------------------------------------------------
// Kernel 2: persistent single-workgroup recurrence.
// 1024 threads = 32 wave32.  Wave w: hj = w&15 (hidden tile), M-tiles
// {2*(w>>4), 2*(w>>4)+1}.  It computes gate tiles (i,f,g,o) at columns
// g*H + hj*16 -> the elementwise LSTM combine is wave-local.
// LDS: h_f32[64x256] + c_f32[64x256] + h_bf16*p[64x256] + lengths.
// ---------------------------------------------------------------------------
extern "C" __global__ __launch_bounds__(1024)
void lstm_recurrent(const float*  __restrict__ G,       // [T*B, NG]
                    const bf16_t* __restrict__ Ubf,     // [NG, D]
                    const long long* __restrict__ lengths, // [B] int64
                    float* __restrict__ out) {          // [T,B,H] ++ h_n ++ c_n
    extern __shared__ char smem[];
    float*  h32 = (float*)smem;                 // 64KB
    float*  c32 = h32 + BB * HH;                // 64KB
    bf16_t* hbf = (bf16_t*)(c32 + BB * HH);     // 32KB  (stores h*PKEEP)
    int*    len = (int*)(hbf + BB * HH);        // 256B

    const int tid   = threadIdx.x;
    const int lane  = tid & 31;
    const int wave  = tid >> 5;
    const int l15   = lane & 15;
    const int lhalf = lane >> 4;
    const int hj    = wave & 15;         // hidden tile 0..15
    const int m0    = (wave >> 4) * 2;   // M-tiles m0, m0+1

    for (int i = tid; i < BB * HH; i += 1024) {
        h32[i] = 0.0f; c32[i] = 0.0f; hbf[i] = (bf16_t)0.0f;
    }
    for (int i = tid; i < BB; i += 1024) len[i] = (int)lengths[i];
    __syncthreads();

    // B-matrix (U) lane base pointers for the 4 gates (L2-resident, 512KB).
    const bf16_t* Bbase[4];
#pragma unroll
    for (int g = 0; g < 4; ++g)
        Bbase[g] = Ubf + (size_t)(g * HH + hj * 16 + l15) * DD + lhalf * 16;

    const int j = hj * 16 + l15;  // this lane's hidden column

    for (int t = 0; t < TT; ++t) {
        // ---- accumulators init from precomputed gate stream G ----
        v8f acc[2][4];
        const size_t grow0 = (size_t)t * BB * NG;
#pragma unroll
        for (int pm = 0; pm < 2; ++pm) {
#pragma unroll
            for (int g = 0; g < 4; ++g) {
                const float* gp = G + grow0 +
                    (size_t)((m0 + pm) * 16 + lhalf * 8) * NG + (g * HH + j);
                v8f a;
#pragma unroll
                for (int r = 0; r < 8; ++r) a[r] = gp[(size_t)r * NG];
                acc[pm][g] = a;
            }
        }

        // Prefetch next step's G slice into the cache hierarchy (CDNA5
        // global_prefetch_b8) while this step computes.
        if (t + 1 < TT) {
            const float* np = G + (size_t)(t + 1) * BB * NG +
                (size_t)(m0 * 16 + lhalf * 8) * NG + j;
            __builtin_prefetch(np, 0, 1);
        }

        // ---- (h * p) @ U^T via WMMA, K = 256 in 8 chunks of 32 ----
        // 8 independent accumulator chains per kc -> no WMMA hazard stalls.
#pragma unroll
        for (int kc = 0; kc < 8; ++kc) {
            v16bf a0, a1;
            const bf16_t* ap = hbf + (size_t)(m0 * 16 + l15) * HH +
                               kc * 32 + lhalf * 16;
            {
                const v8bf x1 = *(const v8bf*)ap;
                const v8bf x2 = *(const v8bf*)(ap + 8);
#pragma unroll
                for (int i = 0; i < 8; ++i) { a0[i] = x1[i]; a0[i + 8] = x2[i]; }
            }
            {
                const v8bf x1 = *(const v8bf*)(ap + 16 * HH);
                const v8bf x2 = *(const v8bf*)(ap + 16 * HH + 8);
#pragma unroll
                for (int i = 0; i < 8; ++i) { a1[i] = x1[i]; a1[i + 8] = x2[i]; }
            }
#pragma unroll
            for (int g = 0; g < 4; ++g) {
                const v16bf bm = *(const v16bf*)(Bbase[g] + kc * 32);
                acc[0][g] = __builtin_amdgcn_wmma_f32_16x16x32_bf16(
                    false, a0, false, bm, (short)0, acc[0][g], false, false);
                acc[1][g] = __builtin_amdgcn_wmma_f32_16x16x32_bf16(
                    false, a1, false, bm, (short)0, acc[1][g], false, false);
            }
        }

        __syncthreads();  // all waves done READING hbf before anyone writes

        // ---- wave-local elementwise LSTM update ----
#pragma unroll
        for (int pm = 0; pm < 2; ++pm) {
#pragma unroll
            for (int r = 0; r < 8; ++r) {
                const int row = (m0 + pm) * 16 + lhalf * 8 + r;
                const float iv = sigf(acc[pm][0][r]);
                const float fv = sigf(acc[pm][1][r]);
                const float gv = tanhf(acc[pm][2][r]);
                const float ov = sigf(acc[pm][3][r]);
                const float cold = c32[row * HH + j];
                const float cnew = fv * cold + iv * gv;
                const float hnew = ov * tanhf(cnew);
                const bool active = (t < len[row]);
                if (active) {
                    c32[row * HH + j] = cnew;
                    h32[row * HH + j] = hnew;
                    hbf[row * HH + j] = (bf16_t)(hnew * PKEEP);
                }
                out[((size_t)t * BB + row) * HH + j] = active ? hnew : 0.0f;
            }
        }
        __syncthreads();  // writes visible before next step's A loads
    }

    // ---- final states h_n, c_n appended after outputs ----
    float* hn = out + (size_t)TT * BB * HH;
    float* cn = hn + BB * HH;
    for (int i = tid; i < BB * HH; i += 1024) { hn[i] = h32[i]; cn[i] = c32[i]; }
}

// ---------------------------------------------------------------------------
// Host launch.  Workspace layout (bytes):
//   Wbf 512KB | Ubf 512KB | Xbf 64MB | G 512MB
// ---------------------------------------------------------------------------
extern "C" void kernel_launch(void* const* d_in, const int* in_sizes, int n_in,
                              void* d_out, int out_size, void* d_ws, size_t ws_size,
                              hipStream_t stream) {
    (void)in_sizes; (void)n_in; (void)out_size; (void)ws_size;
    const float*     x       = (const float*)d_in[0];
    const long long* lengths = (const long long*)d_in[1];
    const float*     W       = (const float*)d_in[2];
    const float*     U       = (const float*)d_in[3];
    const float*     b_ih    = (const float*)d_in[4];
    const float*     b_hh    = (const float*)d_in[5];
    float*           out     = (float*)d_out;

    bf16_t* Wbf = (bf16_t*)d_ws;
    bf16_t* Ubf = Wbf + (size_t)NG * DD;
    bf16_t* Xbf = Ubf + (size_t)NG * DD;
    float*  G   = (float*)(Xbf + (size_t)TT * BB * DD);

    lstm_prep<<<2048, 256, 0, stream>>>(x, W, U, Xbf, Wbf, Ubf);
    lstm_input_proj<<<(TT * BB) / 16, 256, 0, stream>>>(Xbf, Wbf, b_ih, b_hh, G);

    const size_t smem = (size_t)BB * HH * sizeof(float) * 2   // h32, c32
                      + (size_t)BB * HH * sizeof(bf16_t)      // hbf
                      + (size_t)BB * sizeof(int);             // lengths
    lstm_recurrent<<<1, 1024, smem, stream>>>(G, Ubf, lengths, out);
}
